// STDIMModel_88991722373393
// MI455X (gfx1250) — matrix operand order repaired and checked by reading, not compiled, not run on hardware
//
#include <hip/hip_runtime.h>

typedef __bf16 bf16;
typedef __attribute__((ext_vector_type(16))) bf16  v16bf;
typedef __attribute__((ext_vector_type(8)))  float v8f;

#define N_TOK 512
#define YX    256
#define DD    128
#define EE    256
#define BM    64
#define NT    128
#define SCALE (1.0f/131072.0f)   // 1/(Y*X*N)

union FragU { v16bf v; uint4 q[2]; };

// A-operand (16-bit 16x32): lane holds row (lane&15); K(e) = 16*(e>>3) + 8*(lane>>4) + (e&7)
__device__ __forceinline__ v16bf ldfragA(const bf16* row, int kk, int hl) {
  FragU f;
  const bf16* p = row + kk * 32 + 8 * hl;
  f.q[0] = *(const uint4*)(p);
  f.q[1] = *(const uint4*)(p + 16);
  return f.v;
}
// B-operand (16-bit 32x16): lane holds col (lane&15); K(e) = 16*(e>>3) + 8*(lane>>4) + (e&7)
__device__ __forceinline__ v16bf ldfragB(const bf16* row, int kk, int hl) {
  FragU f;
  const bf16* p = row + kk * 32 + 8 * hl;
  f.q[0] = *(const uint4*)(p);
  f.q[1] = *(const uint4*)(p + 16);
  return f.v;
}

__device__ __forceinline__ v8f wmma_bf16(v16bf a, v16bf b, v8f c) {
  return __builtin_amdgcn_wmma_f32_16x16x32_bf16(false, a, false, b, (short)0, c, false, false);
}

// stream fp32 rows (rowStrideF floats apart, DD floats each) -> split hi/lo bf16 in LDS
__device__ __forceinline__ void stage_split(const float* __restrict__ g, long rowStrideF,
                                            bf16* __restrict__ sh, bf16* __restrict__ sl,
                                            int nrows, int tid) {
  const int total = nrows * (DD / 4);
  for (int i = tid; i < total; i += NT) {
    const int row = i >> 5;
    const int q   = (i & 31) << 2;
    const float4 v = *(const float4*)(g + (long)row * rowStrideF + q);
    union { bf16 b[4]; uint2 u; } H, L;
    H.b[0] = (bf16)v.x; H.b[1] = (bf16)v.y; H.b[2] = (bf16)v.z; H.b[3] = (bf16)v.w;
    L.b[0] = (bf16)(v.x - (float)H.b[0]);
    L.b[1] = (bf16)(v.y - (float)H.b[1]);
    L.b[2] = (bf16)(v.z - (float)H.b[2]);
    L.b[3] = (bf16)(v.w - (float)H.b[3]);
    const int o = row * DD + q;
    *(uint2*)(sh + o) = H.u;
    *(uint2*)(sl + o) = L.u;
  }
}

__global__ void zero_out(float* o) { o[0] = 0.0f; }

// pred1 = f_t @ W1^T + b1, stored as split bf16 (hi, lo) in workspace
__global__ __launch_bounds__(128) void pred1_split(
    const float* __restrict__ f_t, const float* __restrict__ W1,
    const float* __restrict__ b1, bf16* __restrict__ ph, bf16* __restrict__ pl) {
  __shared__ float fr[EE];
  const int n = blockIdx.x;
  for (int e = threadIdx.x; e < EE; e += 128) fr[e] = f_t[(long)n * EE + e];
  __syncthreads();
  const int d = threadIdx.x;
  const float* wrow = W1 + (long)d * EE;
  float acc = b1[d];
  #pragma unroll 4
  for (int e = 0; e < EE; ++e) acc = fmaf(fr[e], wrow[e], acc);
  const bf16 h = (bf16)acc;
  ph[(long)n * DD + d] = h;
  pl[(long)n * DD + d] = (bf16)(acc - (float)h);
}

// One block: (y,x) position x 64-row block. 4 waves, 16 rows each.
// Fuses: pred2 GEMM (WMMA, split bf16), both logits GEMMs (WMMA, 3-term split),
// online logsumexp + diagonal capture, InfoNCE reduction -> atomicAdd.
// fmap_tp1 blocks are software-pipelined through 16 float4 staging registers so
// global latency hides under the 96-WMMA compute phase of the previous block.
__global__ __launch_bounds__(NT) void stdim_fused(
    const bf16* __restrict__ p1h, const bf16* __restrict__ p1l,
    const float* __restrict__ fmap_t, const float* __restrict__ fmap_tp1,
    const float* __restrict__ W2, const float* __restrict__ b2,
    float* __restrict__ out) {
  __shared__ __align__(16) bf16 sA2h[BM * DD];           // 16 KB
  __shared__ __align__(16) bf16 sA2l[BM * DD];           // 16 KB
  __shared__ __align__(16) unsigned char sU[98304];      // 96 KB (prologue/stream union)

  const int tid  = threadIdx.x;
  const int lane = tid & 31;
  const int w    = tid >> 5;
  const int m    = lane & 15;
  const int hl   = lane >> 4;
  const int blk  = blockIdx.x;
  const int yx   = blk >> 3;
  const int rb   = (blk & 7) * BM;
  const int wrow = w * 16;
  const int grow = rb + wrow;          // wave's global row base

  bf16* Fth  = (bf16*)sU;              // [BM][DD]
  bf16* Ftl  = Fth + BM * DD;
  bf16* sW2h = Ftl + BM * DD;          // [DD][DD]
  bf16* sW2l = sW2h + DD * DD;
  bf16* Bh   = (bf16*)sU;              // aliases prologue region in main phase
  bf16* Bl   = Bh + BM * DD;

  // per-thread staging coordinates (same mapping for prefetch and store)
  const int srow = tid >> 5;                 // rows 0..3 (+4 per j)
  const int sq   = (tid & 31) << 2;          // float offset within row
  const long fstride = (long)YX * DD;

  // ---- prologue: stage fmap_t rows + W2 (split bf16) ----
  stage_split(fmap_t + ((long)rb * YX + yx) * DD, fstride, Fth, Ftl, BM, tid);
  stage_split(W2, DD, sW2h, sW2l, DD, tid);
  __syncthreads();

  v16bf a1h[4], a1l[4], a2h[4], a2l[4];

  // branch-1 A fragments straight from global workspace (pred1 split)
  {
    const bf16* rh = p1h + (long)(grow + m) * DD;
    const bf16* rl = p1l + (long)(grow + m) * DD;
    #pragma unroll
    for (int kk = 0; kk < 4; ++kk) { a1h[kk] = ldfragA(rh, kk, hl); a1l[kk] = ldfragA(rl, kk, hl); }
  }
  // fmap_t fragments (A operand of the pred2 GEMM); reuse a2 regs
  {
    const bf16* rh = Fth + (wrow + m) * DD;
    const bf16* rl = Ftl + (wrow + m) * DD;
    #pragma unroll
    for (int kk = 0; kk < 4; ++kk) { a2h[kk] = ldfragA(rh, kk, hl); a2l[kk] = ldfragA(rl, kk, hl); }
  }

  // pred2 tile (16 rows x 128 d) via 3-term split WMMA, +b2, re-split into LDS
  for (int ct = 0; ct < 8; ++ct) {
    v8f acc = {0.f, 0.f, 0.f, 0.f, 0.f, 0.f, 0.f, 0.f};
    const bf16* wh = sW2h + (ct * 16 + m) * DD;
    const bf16* wl = sW2l + (ct * 16 + m) * DD;
    #pragma unroll
    for (int kk = 0; kk < 4; ++kk) {
      v16bf bh = ldfragB(wh, kk, hl);
      v16bf bl = ldfragB(wl, kk, hl);
      acc = wmma_bf16(a2h[kk], bh, acc);
      acc = wmma_bf16(a2h[kk], bl, acc);
      acc = wmma_bf16(a2l[kk], bh, acc);
    }
    const int   d  = ct * 16 + m;
    const float bb = b2[d];
    #pragma unroll
    for (int r = 0; r < 8; ++r) {
      const float vv = acc[r] + bb;       // C-layout: row = r + 8*hl, col = m
      const bf16  h  = (bf16)vv;
      const int rowA = wrow + r + 8 * hl;
      sA2h[rowA * DD + d] = h;
      sA2l[rowA * DD + d] = (bf16)(vv - (float)h);
    }
  }
  __syncthreads();

  // branch-2 A fragments from the pred2 tile
  {
    const bf16* rh = sA2h + (wrow + m) * DD;
    const bf16* rl = sA2l + (wrow + m) * DD;
    #pragma unroll
    for (int kk = 0; kk < 4; ++kk) { a2h[kk] = ldfragA(rh, kk, hl); a2l[kk] = ldfragA(rl, kk, hl); }
  }

  // ---- main loop: register-pipelined fmap_tp1 streaming + online logsumexp ----
  float mrow[2][8], srow2[2][8], dg[2] = {0.f, 0.f};
  #pragma unroll
  for (int r = 0; r < 8; ++r) {
    mrow[0][r] = mrow[1][r] = -3.0e38f;
    srow2[0][r] = srow2[1][r] = 0.f;
  }

  float4 st[16];
  {
    const float* g = fmap_tp1 + ((long)0 * YX + yx) * DD;
    #pragma unroll
    for (int j = 0; j < 16; ++j)
      st[j] = *(const float4*)(g + (long)(srow + j * 4) * fstride + sq);
  }

  for (int cb = 0; cb < 8; ++cb) {
    __syncthreads();   // previous block's compute done: safe to overwrite Bh/Bl
    // convert staged registers -> split bf16 LDS
    #pragma unroll
    for (int j = 0; j < 16; ++j) {
      const float4 v = st[j];
      union { bf16 b[4]; uint2 u; } H, L;
      H.b[0] = (bf16)v.x; H.b[1] = (bf16)v.y; H.b[2] = (bf16)v.z; H.b[3] = (bf16)v.w;
      L.b[0] = (bf16)(v.x - (float)H.b[0]);
      L.b[1] = (bf16)(v.y - (float)H.b[1]);
      L.b[2] = (bf16)(v.z - (float)H.b[2]);
      L.b[3] = (bf16)(v.w - (float)H.b[3]);
      const int o = (srow + j * 4) * DD + sq;
      *(uint2*)(Bh + o) = H.u;
      *(uint2*)(Bl + o) = L.u;
    }
    __syncthreads();   // B block ready
    // issue next block's global loads; they complete while we run 96 WMMAs below
    if (cb < 7) {
      const float* g = fmap_tp1 + ((long)((cb + 1) * BM) * YX + yx) * DD;
      #pragma unroll
      for (int j = 0; j < 16; ++j)
        st[j] = *(const float4*)(g + (long)(srow + j * 4) * fstride + sq);
    }

    for (int ct = 0; ct < 4; ++ct) {
      const bf16* brh = Bh + (ct * 16 + m) * DD;
      const bf16* brl = Bl + (ct * 16 + m) * DD;
      v8f acc1 = {0.f, 0.f, 0.f, 0.f, 0.f, 0.f, 0.f, 0.f};
      v8f acc2 = {0.f, 0.f, 0.f, 0.f, 0.f, 0.f, 0.f, 0.f};
      #pragma unroll
      for (int kk = 0; kk < 4; ++kk) {
        v16bf bh = ldfragB(brh, kk, hl);
        v16bf bl = ldfragB(brl, kk, hl);
        acc1 = wmma_bf16(a1h[kk], bh, acc1);
        acc1 = wmma_bf16(a1h[kk], bl, acc1);
        acc1 = wmma_bf16(a1l[kk], bh, acc1);
        acc2 = wmma_bf16(a2h[kk], bh, acc2);
        acc2 = wmma_bf16(a2h[kk], bl, acc2);
        acc2 = wmma_bf16(a2l[kk], bh, acc2);
      }
      // diagonal tile: global col base == wave row base
      if ((cb * BM + ct * 16) == grow && (hl == (m >> 3))) {
        dg[0] = acc1[m & 7];
        dg[1] = acc2[m & 7];
      }
      #pragma unroll
      for (int r = 0; r < 8; ++r) {
        float a  = acc1[r];
        float nm = fmaxf(mrow[0][r], a);
        srow2[0][r] = srow2[0][r] * __expf(mrow[0][r] - nm) + __expf(a - nm);
        mrow[0][r] = nm;
        a  = acc2[r];
        nm = fmaxf(mrow[1][r], a);
        srow2[1][r] = srow2[1][r] * __expf(mrow[1][r] - nm) + __expf(a - nm);
        mrow[1][r] = nm;
      }
    }
  }

  // ---- reduction: per-row logsumexp across 16 lanes, subtract diag, atomic add ----
  float wsum = 0.f;
  #pragma unroll
  for (int b = 0; b < 2; ++b) {
    #pragma unroll
    for (int r = 0; r < 8; ++r) {
      float L = mrow[b][r] + __logf(srow2[b][r]);
      #pragma unroll
      for (int mask = 1; mask < 16; mask <<= 1) {
        const float o  = __shfl_xor(L, mask, 32);
        const float mx = fmaxf(L, o);
        L = mx + __logf(__expf(L - mx) + __expf(o - mx));
      }
      float dv = (m == r + 8 * hl) ? dg[b] : 0.f;
      #pragma unroll
      for (int mask = 1; mask < 16; mask <<= 1) dv += __shfl_xor(dv, mask, 32);
      if (m == 0) wsum += (L - dv);
    }
  }
  wsum += __shfl_xor(wsum, 16, 32);
  if (lane == 0) atomicAdd(out, wsum * SCALE);
}

extern "C" void kernel_launch(void* const* d_in, const int* in_sizes, int n_in,
                              void* d_out, int out_size, void* d_ws, size_t ws_size,
                              hipStream_t stream) {
  (void)in_sizes; (void)n_in; (void)out_size; (void)ws_size;
  const float* f_t      = (const float*)d_in[0];
  const float* fmap_t   = (const float*)d_in[1];
  const float* fmap_tp1 = (const float*)d_in[2];
  const float* W1       = (const float*)d_in[3];
  const float* b1       = (const float*)d_in[4];
  const float* W2       = (const float*)d_in[5];
  const float* b2       = (const float*)d_in[6];
  float* out = (float*)d_out;

  bf16* p1h = (bf16*)d_ws;                       // 512*128 bf16
  bf16* p1l = p1h + (size_t)N_TOK * DD;          // 512*128 bf16 (256 KB total)

  hipLaunchKernelGGL(zero_out, dim3(1), dim3(1), 0, stream, out);
  hipLaunchKernelGGL(pred1_split, dim3(N_TOK), dim3(128), 0, stream, f_t, W1, b1, p1h, p1l);
  hipLaunchKernelGGL(stdim_fused, dim3(YX * (N_TOK / BM)), dim3(NT), 0, stream,
                     p1h, p1l, fmap_t, fmap_tp1, W2, b2, out);
}
